// PPGN_8031588843657
// MI455X (gfx1250) — matrix operand order repaired and checked by reading, not compiled, not run on hardware
//
#include <hip/hip_runtime.h>
#include <hip/hip_bf16.h>

#define NUC 200000
#define NSC 100000
#define NTC 100000
#define NC  (NSC + NUC + NTC)   // 400000
#define DC  32
#define DKC 128
#define METAC 64
#define BC  2048
#define LC  50
#define NNZC 2000000

typedef __attribute__((ext_vector_type(16))) _Float16 v16h;
typedef __attribute__((ext_vector_type(8)))  _Float16 v8h;
typedef __attribute__((ext_vector_type(2)))  _Float16 v2h;
typedef __attribute__((ext_vector_type(8)))  float    v8f;

#define SHUF16(x, y) __builtin_shufflevector(x, y, 0,1,2,3,4,5,6,7,8,9,10,11,12,13,14,15)

__device__ __forceinline__ void atomAddF(float* p, float v) {
    (void)__hip_atomic_fetch_add(p, v, __ATOMIC_RELAXED, __HIP_MEMORY_SCOPE_AGENT);
}

// ---------------------------------------------------------------------------
// Attention pooling: one block (64 threads) per batch sample.
// his[b,d] = sum_l softmax_l( relu(e@w1+b1)@w2 - 1e8*(id==0) ) * e[l,d]
// ---------------------------------------------------------------------------
__global__ void attn_pool_kernel(const int* __restrict__ u,
                                 const int* __restrict__ seq,   // [NU,L]
                                 const float* __restrict__ emb, // [Nitem,32]
                                 const float* __restrict__ w1,  // [32,32]
                                 const float* __restrict__ b1,  // [32]
                                 const float* __restrict__ w2,  // [32]
                                 float* __restrict__ his)       // [B,32]
{
    __shared__ float w1s[32 * 32];
    __shared__ float b1s[32], w2s[32];
    __shared__ int   ids[LC];
    __shared__ float e[LC][32];
    __shared__ float att[LC];
    __shared__ float invsum;

    int tid = threadIdx.x;   // 64 threads
    int b   = blockIdx.x;

    for (int i = tid; i < 32 * 32; i += 64) w1s[i] = w1[i];
    if (tid < 32) { b1s[tid] = b1[tid]; w2s[tid] = w2[tid]; }
    const int* sb = seq + (long long)u[b] * LC;
    if (tid < LC) ids[tid] = sb[tid];
    __syncthreads();

    for (int i = tid; i < LC * 32; i += 64) {
        int l = i >> 5, d = i & 31;
        e[l][d] = emb[(long long)ids[l] * 32 + d];
    }
    __syncthreads();

    for (int l = tid; l < LC; l += 64) {
        float acc = 0.f;
        for (int j = 0; j < 32; ++j) {
            float s = b1s[j];
            #pragma unroll
            for (int d = 0; d < 32; ++d) s += e[l][d] * w1s[d * 32 + j];
            acc += fmaxf(s, 0.f) * w2s[j];
        }
        att[l] = acc - (ids[l] == 0 ? 1e8f : 0.f);
    }
    __syncthreads();

    if (tid == 0) {
        float m = -3.4e38f;
        for (int l = 0; l < LC; ++l) m = fmaxf(m, att[l]);
        float s = 0.f;
        for (int l = 0; l < LC; ++l) { att[l] = __expf(att[l] - m); s += att[l]; }
        invsum = 1.f / s;
    }
    __syncthreads();

    if (tid < 32) {
        float s = 0.f;
        for (int l = 0; l < LC; ++l) s += att[l] * e[l][tid];
        his[b * 32 + tid] = s * invsum;
    }
}

// ---------------------------------------------------------------------------
// h = relu(his @ dec_w1 + dec_b1) : [B,32] x [32,64] -> [B,64]
// ---------------------------------------------------------------------------
__global__ void his_mlp_kernel(const float* __restrict__ his,
                               const float* __restrict__ w1,   // [32,64]
                               const float* __restrict__ b1,   // [64]
                               float* __restrict__ h)          // [B,64]
{
    int tid = blockIdx.x * blockDim.x + threadIdx.x;
    if (tid >= BC * METAC) return;
    int b = tid >> 6, j = tid & 63;
    float s = b1[j];
    const float* hb = his + b * 32;
    #pragma unroll
    for (int d = 0; d < 32; ++d) s += hb[d] * w1[d * METAC + j];
    h[tid] = fmaxf(s, 0.f);
}

// ---------------------------------------------------------------------------
// SpMM hop: one wave per nnz, lane = embedding channel.  Indices are wave-
// uniform -> readfirstlane makes them SGPR loads; the per-lane work is one
// coalesced 128B gather + one coalesced 128B GLOBAL_ATOMIC_ADD_F32 scatter,
// both L2-resident (each 51MB layer fits the 192MB L2).
// src==null means layer-0 (read from the 3 concatenated tables).
// ---------------------------------------------------------------------------
__global__ void spmm_kernel(const int* __restrict__ rows,
                            const int* __restrict__ cols,
                            const float* __restrict__ vals,
                            const float* __restrict__ src,
                            const float* __restrict__ semb,
                            const float* __restrict__ uemb,
                            const float* __restrict__ temb,
                            float* __restrict__ dst, int nnz)
{
    int i = __builtin_amdgcn_readfirstlane(blockIdx.x * 8 + (threadIdx.x >> 5));
    if (i >= nnz) return;
    int lane = threadIdx.x & 31;
    int r = rows[i], c = cols[i];
    float v = vals[i];
    float x;
    if (src) {
        x = src[(long long)c * 32 + lane];
    } else {
        if (c < NSC)            x = semb[(long long)c * 32 + lane];
        else if (c < NSC + NUC) x = uemb[(long long)(c - NSC) * 32 + lane];
        else                    x = temb[(long long)(c - NSC - NUC) * 32 + lane];
    }
    atomAddF(dst + (long long)r * 32 + lane, v * x);
}

// ---------------------------------------------------------------------------
// Gather layer-k slice of the [N,DK] feature for si / u / ti rows.
// tid layout: which(3) x b(2048) x d(32);  B*32 == 65536 so which = tid>>16.
// ---------------------------------------------------------------------------
__global__ void gather_kernel(const float* __restrict__ src,   // null -> tables
                              const float* __restrict__ semb,
                              const float* __restrict__ uemb,
                              const float* __restrict__ temb,
                              int layer,
                              const int* __restrict__ si,
                              const int* __restrict__ u,
                              const int* __restrict__ ti,
                              float* __restrict__ featS,
                              float* __restrict__ featU,
                              float* __restrict__ featT)
{
    int tid = blockIdx.x * blockDim.x + threadIdx.x;
    if (tid >= 3 * BC * 32) return;
    int d     = tid & 31;
    int b     = (tid >> 5) & (BC - 1);
    int which = tid >> 16;
    int row = (which == 0) ? si[b] : (which == 1) ? (NSC + u[b]) : (NSC + NUC + ti[b]);
    float x;
    if (src) {
        x = src[(long long)row * 32 + d];
    } else {
        if (row < NSC)            x = semb[(long long)row * 32 + d];
        else if (row < NSC + NUC) x = uemb[(long long)(row - NSC) * 32 + d];
        else                      x = temb[(long long)(row - NSC - NUC) * 32 + d];
    }
    float* dstf = (which == 0) ? featS : (which == 1) ? featU : featT;
    dstf[b * DKC + layer * 32 + d] = x;
}

// ---------------------------------------------------------------------------
// Meta-decode GEMM via WMMA:
//   u_meta[b,e] = sum_{m,d} (h[b,m]*ue[b,d]) * w2[m, d*128+e]
//              +  sum_{d}    ue[b,d]         * b2[d*128+e]
// [2048 x 8320] @ [8320 x 128]; A generated on the fly as outer(h, ue).
// Block = 256 threads (8 waves), 32-row M tile; wave owns one 16-col N tile
// and issues TWO independent WMMAs per K-chunk sharing one B fragment.
// Fragments pack with b128 LDS loads (A row-contiguous; B staged TRANSPOSED
// BtT[e][k] so its per-lane run is contiguous).  Rows padded to 40 halves
// (80B) to keep 16B alignment.  Staging is vectorized: float4 global loads
// along e, (j,j+1) pairs packed into one 32-bit LDS store.
// gridDim.y selects the domain (source / target) so both run in one launch.
// ---------------------------------------------------------------------------
__global__ void meta_gemm_kernel(const float* __restrict__ h0,
                                 const float* __restrict__ h1,
                                 const float* __restrict__ ue,   // [B,128]
                                 const float* __restrict__ w2a,  // [64,16384]
                                 const float* __restrict__ w2b,
                                 const float* __restrict__ b2a,  // [16384]
                                 const float* __restrict__ b2b,
                                 float* __restrict__ out0,       // [B,128]
                                 float* __restrict__ out1)
{
    const float* h  = blockIdx.y ? h1 : h0;
    const float* w2 = blockIdx.y ? w2b : w2a;
    const float* b2 = blockIdx.y ? b2b : b2a;
    float* out      = blockIdx.y ? out1 : out0;

    __shared__ float    hS[32][METAC];     // 8 KB
    __shared__ float    ueS[32][DKC];      // 16 KB
    __shared__ _Float16 At[32][40];        // 2.5 KB (padded rows, 80B stride)
    __shared__ _Float16 BtT[DKC][40];      // 10 KB (transposed, padded)

    int tid  = threadIdx.x;     // 256
    int wave = tid >> 5;        // 0..7
    int lane = tid & 31;
    int brow = blockIdx.x * 32;

    for (int i = tid; i < 32 * METAC; i += 256)
        hS[i >> 6][i & 63] = h[(brow + (i >> 6)) * METAC + (i & 63)];
    for (int i = tid; i < 32 * DKC; i += 256)
        ueS[i >> 7][i & 127] = ue[(brow + (i >> 7)) * DKC + (i & 127)];
    __syncthreads();

    v8f acc0 = {}, acc1 = {};
    int hf   = lane >> 4;
    int r16  = lane & 15;
    int ecol = wave * 16;
    int crot = (lane >> 2) & 3;   // store-order rotation to spread LDS banks

    for (int kc = 0; kc < 260; ++kc) {
        const float* bsrc;
        int m = 0, db;
        if (kc < 256) { m = kc >> 2; db = (kc & 3) << 5; bsrc = w2 + m * 16384 + db * 128; }
        else          { db = (kc - 256) << 5;            bsrc = b2 + db * 128; }

        // ---- stage B chunk (32 x 128) transposed ----
        // g: e0 = (g&31)*4 (fast, coalesced float4 loads), j0 = (g>>5)*2
        for (int g = tid; g < 512; g += 256) {
            int e0 = (g & 31) << 2;
            int j0 = (g >> 5) << 1;
            float4 f0 = *(const float4*)&bsrc[j0 * 128 + e0];
            float4 f1 = *(const float4*)&bsrc[(j0 + 1) * 128 + e0];
            float a0[4] = {f0.x, f0.y, f0.z, f0.w};
            float a1[4] = {f1.x, f1.y, f1.z, f1.w};
            #pragma unroll
            for (int t = 0; t < 4; ++t) {
                int c = (t + crot) & 3;
                v2h p = {(_Float16)a0[c], (_Float16)a1[c]};
                *(v2h*)&BtT[e0 + c][j0] = p;   // 32-bit LDS store
            }
        }
        // ---- stage A chunk (32 x 32) = h[:,m] (x) ue[:, db:db+32] ----
        for (int g = tid; g < 512; g += 256) {
            int rr = g >> 4;
            int jj = (g & 15) << 1;
            float s = (kc < 256) ? hS[rr][m] : 1.0f;
            v2h p = {(_Float16)(s * ueS[rr][db + jj]),
                     (_Float16)(s * ueS[rr][db + jj + 1])};
            *(v2h*)&At[rr][jj] = p;
        }
        __syncthreads();

        // B fragment: two b128 loads, k = hf*8.. and 16+hf*8.. in BtT row
        v8h blo = *(const v8h*)&BtT[ecol + r16][hf * 8];
        v8h bhi = *(const v8h*)&BtT[ecol + r16][16 + hf * 8];
        v16h bf = SHUF16(blo, bhi);
        // A fragments (rows 0-15 and 16-31): two b128 loads each
        v8h a00 = *(const v8h*)&At[r16][hf * 8];
        v8h a01 = *(const v8h*)&At[r16][16 + hf * 8];
        v16h a0 = SHUF16(a00, a01);
        v8h a10 = *(const v8h*)&At[16 + r16][hf * 8];
        v8h a11 = *(const v8h*)&At[16 + r16][16 + hf * 8];
        v16h a1 = SHUF16(a10, a11);

        acc0 = __builtin_amdgcn_wmma_f32_16x16x32_f16(
                   false, a0, false, bf, (short)0, acc0, false, false);
        acc1 = __builtin_amdgcn_wmma_f32_16x16x32_f16(
                   false, a1, false, bf, (short)0, acc1, false, false);
        __syncthreads();
    }

    // C/D layout: vgpr v, lanes 0-15 -> M=v, lanes 16-31 -> M=8+v, N=lane%16
    #pragma unroll
    for (int v = 0; v < 8; ++v) {
        int Mr = v + (hf ? 8 : 0);
        out[(brow + Mr) * DKC + ecol + r16]      = acc0[v];
        out[(brow + 16 + Mr) * DKC + ecol + r16] = acc1[v];
    }
}

// ---------------------------------------------------------------------------
// NCF head: x=[u_meta|feat] (256) -> relu 128 -> relu 64 -> sigmoid.
// One block (128 threads) per sample.
// ---------------------------------------------------------------------------
__global__ void ncf_kernel(const float* __restrict__ umeta,  // [B,128]
                           const float* __restrict__ feat,   // [B,128]
                           const float* __restrict__ w1,     // [256,128]
                           const float* __restrict__ b1,     // [128]
                           const float* __restrict__ w2,     // [128,64]
                           const float* __restrict__ b2,     // [64]
                           const float* __restrict__ dw,     // [64]
                           const float* __restrict__ db,     // [1]
                           float* __restrict__ out)          // [B]
{
    __shared__ float x[256];
    __shared__ float h1[128];
    __shared__ float h2[64];
    int tid = threadIdx.x;   // 128
    int b   = blockIdx.x;
    x[tid]       = umeta[b * DKC + tid];
    x[128 + tid] = feat[b * DKC + tid];
    __syncthreads();

    float s = b1[tid];
    for (int k = 0; k < 256; ++k) s += x[k] * w1[k * 128 + tid];
    h1[tid] = fmaxf(s, 0.f);
    __syncthreads();

    if (tid < 64) {
        float s2 = b2[tid];
        for (int k = 0; k < 128; ++k) s2 += h1[k] * w2[k * 64 + tid];
        h2[tid] = fmaxf(s2, 0.f);
    }
    __syncthreads();

    if (tid == 0) {
        float s3 = db[0];
        for (int k = 0; k < 64; ++k) s3 += h2[k] * dw[k];
        out[b] = 1.f / (1.f + __expf(-s3));
    }
}

// ---------------------------------------------------------------------------
extern "C" void kernel_launch(void* const* d_in, const int* in_sizes, int n_in,
                              void* d_out, int out_size, void* d_ws, size_t ws_size,
                              hipStream_t stream)
{
    (void)in_sizes; (void)n_in; (void)out_size; (void)ws_size;

    const int*   u        = (const int*)d_in[0];
    const int*   si       = (const int*)d_in[1];
    const int*   ti       = (const int*)d_in[2];
    const int*   src_seq  = (const int*)d_in[3];
    const int*   tgt_seq  = (const int*)d_in[4];
    const int*   adj_rows = (const int*)d_in[5];
    const int*   adj_cols = (const int*)d_in[6];
    const float* adj_vals = (const float*)d_in[7];
    const float* user_emb = (const float*)d_in[8];
    const float* s_emb    = (const float*)d_in[9];
    const float* t_emb    = (const float*)d_in[10];
    const float* a_s_w1   = (const float*)d_in[11];
    const float* a_s_b1   = (const float*)d_in[12];
    const float* a_s_w2   = (const float*)d_in[13];
    const float* a_t_w1   = (const float*)d_in[14];
    const float* a_t_b1   = (const float*)d_in[15];
    const float* a_t_w2   = (const float*)d_in[16];
    const float* d_ss_w1  = (const float*)d_in[17];
    const float* d_ss_b1  = (const float*)d_in[18];
    const float* d_ss_w2  = (const float*)d_in[19];
    const float* d_ss_b2  = (const float*)d_in[20];
    const float* d_tt_w1  = (const float*)d_in[21];
    const float* d_tt_b1  = (const float*)d_in[22];
    const float* d_tt_w2  = (const float*)d_in[23];
    const float* d_tt_b2  = (const float*)d_in[24];
    const float* m_s_w1   = (const float*)d_in[25];
    const float* m_s_b1   = (const float*)d_in[26];
    const float* m_s_w2   = (const float*)d_in[27];
    const float* m_s_b2   = (const float*)d_in[28];
    const float* de_s_w   = (const float*)d_in[29];
    const float* de_s_b   = (const float*)d_in[30];
    const float* m_t_w1   = (const float*)d_in[31];
    const float* m_t_b1   = (const float*)d_in[32];
    const float* m_t_w2   = (const float*)d_in[33];
    const float* m_t_b2   = (const float*)d_in[34];
    const float* de_t_w   = (const float*)d_in[35];
    const float* de_t_b   = (const float*)d_in[36];

    float* out = (float*)d_out;

    // ---- workspace carve-up (all 256B aligned) ----
    char* ws = (char*)d_ws;
    size_t off = 0;
    auto take = [&](size_t bytes) -> float* {
        float* p = (float*)(ws + off);
        off += (bytes + 255) & ~(size_t)255;
        return p;
    };
    const size_t embBytes = (size_t)NC * 32 * sizeof(float);   // 51.2 MB
    float* embA  = take(embBytes);
    float* embB  = take(embBytes);
    float* featS = take((size_t)BC * DKC * sizeof(float));
    float* featU = take((size_t)BC * DKC * sizeof(float));
    float* featT = take((size_t)BC * DKC * sizeof(float));
    float* hisS  = take((size_t)BC * DC * sizeof(float));
    float* hisT  = take((size_t)BC * DC * sizeof(float));
    float* hSbuf = take((size_t)BC * METAC * sizeof(float));
    float* hTbuf = take((size_t)BC * METAC * sizeof(float));
    float* umS   = take((size_t)BC * DKC * sizeof(float));
    float* umT   = take((size_t)BC * DKC * sizeof(float));

    const int spmmBlocks   = (NNZC + 7) / 8;   // 8 waves (8 nnz) per block
    const int gatherBlocks = (3 * BC * 32 + 255) / 256;

    // ---- attention pooling + meta MLP first layer ----
    attn_pool_kernel<<<BC, 64, 0, stream>>>(u, src_seq, s_emb, a_s_w1, a_s_b1, a_s_w2, hisS);
    attn_pool_kernel<<<BC, 64, 0, stream>>>(u, tgt_seq, t_emb, a_t_w1, a_t_b1, a_t_w2, hisT);
    his_mlp_kernel<<<(BC * METAC + 255) / 256, 256, 0, stream>>>(hisS, d_ss_w1, d_ss_b1, hSbuf);
    his_mlp_kernel<<<(BC * METAC + 255) / 256, 256, 0, stream>>>(hisT, d_tt_w1, d_tt_b1, hTbuf);

    // ---- graph encoder: layer 0 gather + 3 SpMM hops (ping-pong buffers) ----
    gather_kernel<<<gatherBlocks, 256, 0, stream>>>(nullptr, s_emb, user_emb, t_emb, 0,
                                                    si, u, ti, featS, featU, featT);
    hipMemsetAsync(embA, 0, embBytes, stream);
    spmm_kernel<<<spmmBlocks, 256, 0, stream>>>(adj_rows, adj_cols, adj_vals, nullptr,
                                                s_emb, user_emb, t_emb, embA, NNZC);
    gather_kernel<<<gatherBlocks, 256, 0, stream>>>(embA, nullptr, nullptr, nullptr, 1,
                                                    si, u, ti, featS, featU, featT);
    hipMemsetAsync(embB, 0, embBytes, stream);
    spmm_kernel<<<spmmBlocks, 256, 0, stream>>>(adj_rows, adj_cols, adj_vals, embA,
                                                nullptr, nullptr, nullptr, embB, NNZC);
    gather_kernel<<<gatherBlocks, 256, 0, stream>>>(embB, nullptr, nullptr, nullptr, 2,
                                                    si, u, ti, featS, featU, featT);
    hipMemsetAsync(embA, 0, embBytes, stream);
    spmm_kernel<<<spmmBlocks, 256, 0, stream>>>(adj_rows, adj_cols, adj_vals, embB,
                                                nullptr, nullptr, nullptr, embA, NNZC);
    gather_kernel<<<gatherBlocks, 256, 0, stream>>>(embA, nullptr, nullptr, nullptr, 3,
                                                    si, u, ti, featS, featU, featT);

    // ---- meta-decode WMMA GEMMs: both domains in one launch (gridDim.y=2) ----
    meta_gemm_kernel<<<dim3(BC / 32, 2), 256, 0, stream>>>(
        hSbuf, hTbuf, featU, d_ss_w2, d_tt_w2, d_ss_b2, d_tt_b2, umS, umT);

    // ---- NCF heads ----
    ncf_kernel<<<BC, 128, 0, stream>>>(umS, featS, m_s_w1, m_s_b1, m_s_w2, m_s_b2,
                                       de_s_w, de_s_b, out);
    ncf_kernel<<<BC, 128, 0, stream>>>(umT, featT, m_t_w1, m_t_b1, m_t_w2, m_t_b2,
                                       de_t_w, de_t_b, out + BC);
}
